// GeometricResonantStateMemory_2714419331740
// MI455X (gfx1250) — compile-verified, matrix operand order checked by reading, not run on hardware
//
#include <hip/hip_runtime.h>

// ---------------- problem constants (match reference) ----------------
#define Dm   1024
#define Sn   32
#define Bn   2
#define Tn   2048
#define Cn   128
#define NC   (Tn / Cn)          // 16 chunks per batch
#define ALPHA 0.05f
#define Af    0.95f             // 1 - alpha
#define SCALE 0.03125f          // 1/sqrt(1024)

typedef __attribute__((ext_vector_type(16))) __bf16 bf16x16;
typedef __attribute__((ext_vector_type(8)))  float  f32x8;
typedef __attribute__((ext_vector_type(4)))  unsigned int u32x4;
typedef __attribute__((ext_vector_type(8)))  int    i32x8;
typedef __attribute__((ext_vector_type(4)))  int    i32x4;

#if __has_builtin(__builtin_amdgcn_tensor_load_to_lds)
#define HAVE_TDM 1
#else
#define HAVE_TDM 0
#endif

struct Frag32B { uint4 lo, hi; };

__device__ __forceinline__ unsigned short f2bf(float f) {
    unsigned int u = __builtin_bit_cast(unsigned int, f);
    u += 0x7fffu + ((u >> 16) & 1u);          // round-to-nearest-even
    return (unsigned short)(u >> 16);
}

__device__ __forceinline__ float wave_sum(float x) {
    #pragma unroll
    for (int off = 16; off >= 1; off >>= 1) x += __shfl_xor(x, off, 32);
    return x;
}
__device__ __forceinline__ float wave_max(float x) {
    #pragma unroll
    for (int off = 16; off >= 1; off >>= 1) x = fmaxf(x, __shfl_xor(x, off, 32));
    return x;
}

#if HAVE_TDM
__device__ __forceinline__ void tdm_call(u32x4 g0, i32x8 g1) {
    i32x4 z4 = {0, 0, 0, 0};
#if defined(__clang_major__) && (__clang_major__ >= 23)
    i32x8 z8 = {0, 0, 0, 0, 0, 0, 0, 0};
    __builtin_amdgcn_tensor_load_to_lds(g0, g1, z4, z4, z8, 0);
#else
    __builtin_amdgcn_tensor_load_to_lds(g0, g1, z4, z4, 0);
#endif
}

// 128-row x 32-elem bf16 tile, row stride 1024 elems, LDS pad 4 DW after each 16 DW
// (reproduces the 40-half / 80-byte LDS pitch used by the fragment loader).
__device__ __forceinline__ void tdm_issue_bf16_tile(unsigned ldsOff,
                                                    const unsigned short* gp,
                                                    unsigned rows) {
    unsigned long long ga = (unsigned long long)gp;
    u32x4 g0;
    g0[0] = 1u;                                   // count=1, user descriptor
    g0[1] = ldsOff;                               // lds_addr (bytes)
    g0[2] = (unsigned)ga;                         // global_addr[31:0]
    g0[3] = (unsigned)((ga >> 32) & 0x01FFFFFFu) | 0x80000000u;  // addr[56:32] | type=2
    i32x8 g1;
    g1[0] = (int)((1u << 16) | (1u << 20) | (3u << 22) | (3u << 25)); // 2B elems, pad 4DW/16DW
    g1[1] = (int)((1024u & 0xFFFFu) << 16);       // tensor_dim0 = 1024 (lo16)
    g1[2] = (int)((1024u >> 16) | ((rows & 0xFFFFu) << 16));          // dim0 hi | dim1 lo
    g1[3] = (int)((rows >> 16) | (32u << 16));    // dim1 hi | tile_dim0 = 32
    g1[4] = (int)128u;                            // tile_dim1 = 128, tile_dim2 = 0
    g1[5] = (int)1024;                            // tensor_dim0_stride lo
    g1[6] = 0;
    g1[7] = 0;
    tdm_call(g0, g1);
}

// contiguous 1-D fp32 copy (elems <= 65535) to LDS offset
__device__ __forceinline__ void tdm_issue_f32_1d(unsigned ldsOff,
                                                 const float* gp, unsigned elems) {
    unsigned long long ga = (unsigned long long)gp;
    u32x4 g0;
    g0[0] = 1u;
    g0[1] = ldsOff;
    g0[2] = (unsigned)ga;
    g0[3] = (unsigned)((ga >> 32) & 0x01FFFFFFu) | 0x80000000u;
    i32x8 g1;
    g1[0] = (int)(2u << 16);                      // data_size = 4B, no pad
    g1[1] = (int)((elems & 0xFFFFu) << 16);       // tensor_dim0 lo
    g1[2] = (int)((elems >> 16) | (1u << 16));    // dim0 hi | tensor_dim1 = 1
    g1[3] = (int)((elems & 0xFFFFu) << 16);       // tile_dim0 = elems
    g1[4] = (int)1u;                              // tile_dim1 = 1
    g1[5] = (int)elems;                           // stride (unused for 1 row)
    g1[6] = 0;
    g1[7] = 0;
    tdm_call(g0, g1);
}
#endif // HAVE_TDM

// ---------------- K1: fp32 -> bf16 weight cast (Wq, Wv) ----------------
__global__ __launch_bounds__(256)
void cast_weights(const float* __restrict__ wq, const float* __restrict__ wv,
                  unsigned short* __restrict__ wq16, unsigned short* __restrict__ wv16) {
    const float* src = blockIdx.y ? wv : wq;
    unsigned short* dst = blockIdx.y ? wv16 : wq16;
    int i = (blockIdx.x * 256 + threadIdx.x) * 4;   // grid.x = D*D/1024
    float4 x = *(const float4*)(src + i);
    *(ushort4*)(dst + i) = make_ushort4(f2bf(x.x), f2bf(x.y), f2bf(x.z), f2bf(x.w));
}

// ---------------- K2: fused LayerNorm + bf16 cast of h ----------------
__global__ __launch_bounds__(256)
void ln_cast(const float* __restrict__ h, const float* __restrict__ gamma,
             const float* __restrict__ beta, unsigned short* __restrict__ hn) {
    int row = blockIdx.x;               // token index in [0, B*T)
    int tid = threadIdx.x;
    float4 x = ((const float4*)(h + (size_t)row * Dm))[tid];
    float s  = x.x + x.y + x.z + x.w;
    float s2 = x.x*x.x + x.y*x.y + x.z*x.z + x.w*x.w;
    __shared__ float red[8][2];
    s  = wave_sum(s);
    s2 = wave_sum(s2);
    int wave = tid >> 5, lane = tid & 31;
    if (lane == 0) { red[wave][0] = s; red[wave][1] = s2; }
    __syncthreads();
    __shared__ float stats[2];
    if (wave == 0) {
        float a = (lane < 8) ? red[lane][0] : 0.f;
        float b = (lane < 8) ? red[lane][1] : 0.f;
        a = wave_sum(a); b = wave_sum(b);
        if (lane == 0) {
            float mu  = a * (1.0f / Dm);
            float var = b * (1.0f / Dm) - mu * mu;
            stats[0] = mu;
            stats[1] = rsqrtf(var + 1e-5f);
        }
    }
    __syncthreads();
    float mu = stats[0], rstd = stats[1];
    int d = tid * 4;
    float4 g = *(const float4*)(gamma + d);
    float4 bt = *(const float4*)(beta + d);
    float y0 = (x.x - mu) * rstd * g.x + bt.x;
    float y1 = (x.y - mu) * rstd * g.y + bt.y;
    float y2 = (x.z - mu) * rstd * g.z + bt.z;
    float y3 = (x.w - mu) * rstd * g.w + bt.w;
    ((ushort4*)(hn + (size_t)row * Dm))[tid] = make_ushort4(f2bf(y0), f2bf(y1), f2bf(y2), f2bf(y3));
}

// ---------------- K3: WMMA bf16 GEMM: q = hn @ Wq^T, v = hn @ Wv^T ----------------
// A tile layout (ISA 05_wmma): lanes 0-15 hold row M=lr with K {k..k+7, k+16..k+23},
// lanes 16-31 hold K {k+8..k+15, k+24..k+31}. B (= W^T) mirrors with column = lr.
#define LDSPITCH 40            // halfs; 80B row pitch (16B-aligned, bank-skewed)
#define TILEBYTES (128 * LDSPITCH * 2)   // 10240 B per staged tile
#define BUFBYTES  (2 * TILEBYTES)        // A + B per buffer

__device__ __forceinline__ bf16x16 load_frag(const unsigned short* lds, int row, int half) {
    const uint4* p0 = (const uint4*)(lds + row * LDSPITCH + half * 8);
    const uint4* p1 = (const uint4*)(lds + row * LDSPITCH + 16 + half * 8);
    Frag32B f; f.lo = *p0; f.hi = *p1;
    return __builtin_bit_cast(bf16x16, f);
}

__global__ __launch_bounds__(256)
void gemm_qv(const unsigned short* __restrict__ hn,
             const unsigned short* __restrict__ wq16, const unsigned short* __restrict__ wv16,
             const float* __restrict__ bq, const float* __restrict__ bv,
             float* __restrict__ qout, float* __restrict__ vout) {
    extern __shared__ __align__(16) unsigned short smem_us[];  // 2 x (A tile + B tile)

    const unsigned short* W = blockIdx.z ? wv16 : wq16;
    const float* bias       = blockIdx.z ? bv : bq;
    float* out              = blockIdx.z ? vout : qout;

    const int mBase = blockIdx.x * 128;     // M = B*T = 4096
    const int nBase = blockIdx.y * 128;     // N = D = 1024
    const int tid = threadIdx.x;
    const int wave = tid >> 5, lane = tid & 31;
    const int wm = (wave >> 2) * 64;        // 2x4 wave grid, each wave 64x32
    const int wn = (wave & 3) * 32;
    const int half = lane >> 4, lr = lane & 15;

    f32x8 acc[4][2] = {};

#if HAVE_TDM
    // ---- TDM double-buffered pipeline: DMA engine fills LDS, waves do WMMA ----
    const bool waveLeader = (__builtin_amdgcn_readfirstlane(wave) == 0); // scalar branch:
    // TDM ignores EXEC, so the issue must be skipped by *branch*, not lane masking.
    if (waveLeader) {
        tdm_issue_bf16_tile(0u,         hn + (size_t)mBase * Dm, 4096u);
        tdm_issue_bf16_tile(TILEBYTES,  W  + (size_t)nBase * Dm, 1024u);
    }
    int cur = 0;
    for (int k0 = 0; k0 < Dm; k0 += 32) {
        if (waveLeader) __builtin_amdgcn_s_wait_tensorcnt(0);
        __syncthreads();                       // publish buf[cur]; all readers of buf[cur^1] done
        if (waveLeader && (k0 + 32 < Dm)) {    // prefetch next K slab into the other buffer
            tdm_issue_bf16_tile((unsigned)((cur ^ 1) * BUFBYTES),
                                hn + (size_t)mBase * Dm + k0 + 32, 4096u);
            tdm_issue_bf16_tile((unsigned)((cur ^ 1) * BUFBYTES + TILEBYTES),
                                W + (size_t)nBase * Dm + k0 + 32, 1024u);
        }
        const unsigned short* curA = smem_us + cur * (BUFBYTES / 2);   // offsets in halfs
        const unsigned short* curB = curA + (TILEBYTES / 2);

        bf16x16 bf[2];
        #pragma unroll
        for (int j = 0; j < 2; ++j) bf[j] = load_frag(curB, wn + j * 16 + lr, half);
        #pragma unroll
        for (int i = 0; i < 4; ++i) {
            bf16x16 af = load_frag(curA, wm + i * 16 + lr, half);
            #pragma unroll
            for (int j = 0; j < 2; ++j) {
                acc[i][j] = __builtin_amdgcn_wmma_f32_16x16x32_bf16(
                    false, af, false, bf[j], (short)0, acc[i][j], false, false);
            }
        }
        cur ^= 1;
    }
#else
    // ---- fallback: VGPR staging ----
    const int row = tid >> 1;
    const int seg = (tid & 1) * 16;
    unsigned short* lA = smem_us;
    unsigned short* lB = smem_us + TILEBYTES / 2;
    for (int k0 = 0; k0 < Dm; k0 += 32) {
        const uint4* gA = (const uint4*)(hn + (size_t)(mBase + row) * Dm + k0 + seg);
        const uint4* gB = (const uint4*)(W  + (size_t)(nBase + row) * Dm + k0 + seg);
        uint4 a0 = gA[0], a1 = gA[1];
        uint4 b0 = gB[0], b1 = gB[1];
        uint4* sA = (uint4*)(lA + row * LDSPITCH + seg);
        uint4* sB = (uint4*)(lB + row * LDSPITCH + seg);
        sA[0] = a0; sA[1] = a1;
        sB[0] = b0; sB[1] = b1;
        __syncthreads();
        bf16x16 bf[2];
        #pragma unroll
        for (int j = 0; j < 2; ++j) bf[j] = load_frag(lB, wn + j * 16 + lr, half);
        #pragma unroll
        for (int i = 0; i < 4; ++i) {
            bf16x16 af = load_frag(lA, wm + i * 16 + lr, half);
            #pragma unroll
            for (int j = 0; j < 2; ++j) {
                acc[i][j] = __builtin_amdgcn_wmma_f32_16x16x32_bf16(
                    false, af, false, bf[j], (short)0, acc[i][j], false, false);
            }
        }
        __syncthreads();
    }
#endif

    // epilogue: C layout — VGPR r: lanes0-15 row r, lanes16-31 row r+8; col = lr
    #pragma unroll
    for (int i = 0; i < 4; ++i) {
        #pragma unroll
        for (int j = 0; j < 2; ++j) {
            int tm = mBase + wm + i * 16;
            int tn = nBase + wn + j * 16 + lr;
            float bb = bias[tn];
            #pragma unroll
            for (int r = 0; r < 8; ++r)
                out[(size_t)(tm + r + half * 8) * Dm + tn] = acc[i][j][r] + bb;
        }
    }
}

// ---------------- K4: routing softmax w = softmax(q . state0 * scale) ----------------
__global__ __launch_bounds__(128)
void routing(const float* __restrict__ q, const float* __restrict__ state0,
             float* __restrict__ wrt) {
    int g = blockIdx.x;                 // token in [0, B*T)
    int b = g / Tn;
    int tid = threadIdx.x, wave = tid >> 5, lane = tid & 31;
    __shared__ __align__(16) float qs[Dm];
    __shared__ float sc[Sn];
    #pragma unroll
    for (int r = 0; r < 2; ++r) {
        int idx = r * 128 + tid;        // 256 float4 total
        ((float4*)qs)[idx] = ((const float4*)(q + (size_t)g * Dm))[idx];
    }
    __syncthreads();
    #pragma unroll
    for (int si = 0; si < 8; ++si) {
        int s = wave * 8 + si;
        const float* st = state0 + (size_t)(b * Sn + s) * Dm;
        float p = 0.f;
        for (int d = lane; d < Dm; d += 32) p += qs[d] * st[d];
        p = wave_sum(p);
        if (lane == 0) sc[s] = p * SCALE;
    }
    __syncthreads();
    if (tid < 32) {
        float x = sc[tid];
        float m = wave_max(x);
        float e = __expf(x - m);
        float sum = wave_sum(e);
        wrt[(size_t)g * Sn + tid] = e / sum;
    }
}

// ---------------- K5a: per-chunk decayed local sum (register accumulators) ----------------
__global__ __launch_bounds__(1024)
void chunk_local(const float* __restrict__ v, const float* __restrict__ wrt,
                 float* __restrict__ local) {
    int b = blockIdx.x >> 4, c = blockIdx.x & 15;
    int tid = threadIdx.x;
    int s = tid >> 5;                   // slot
    int dbase = (tid & 31) * 32;        // 32 d-elements per thread
    float acc[32];
    #pragma unroll
    for (int j = 0; j < 32; ++j) acc[j] = 0.f;
    __shared__ float vrow[Dm];
    __shared__ float wr[Sn];
    for (int i = 0; i < Cn; ++i) {
        int g = b * Tn + c * Cn + i;
        vrow[tid] = v[(size_t)g * Dm + tid];
        if (tid < Sn) wr[tid] = wrt[(size_t)g * Sn + tid];
        __syncthreads();
        float uw = ALPHA * wr[s];
        #pragma unroll
        for (int j = 0; j < 32; ++j) acc[j] = Af * acc[j] + uw * vrow[dbase + j];
        __syncthreads();
    }
    float* dst = local + ((size_t)(b * NC + c) * Sn + s) * Dm + dbase;
    #pragma unroll
    for (int j = 0; j < 32; ++j) dst[j] = acc[j];
}

// ---------------- K5b: 16-step carry scan -> chunk entry states + final_state ----------------
__global__ __launch_bounds__(256)
void chunk_scan(const float* __restrict__ state0, const float* __restrict__ local,
                float* __restrict__ entry, float* __restrict__ finalState) {
    int b = blockIdx.x >> 7;
    int e = (blockIdx.x & 127) * 256 + threadIdx.x;     // element in [0, S*D)
    const float A128 = __powf(Af, (float)Cn);
    float st = state0[(size_t)b * Sn * Dm + e];
    for (int c = 0; c < NC; ++c) {
        size_t off = (size_t)(b * NC + c) * Sn * Dm + e;
        entry[off] = st;
        st = A128 * st + local[off];
    }
    finalState[(size_t)b * Sn * Dm + e] = st;
}

// ---------------- K6: fused per-chunk state replay + slot-attention read ----------------
// LDS-resident (S,D) state; never materializes (B,T,S,D) = 537 MB.
__global__ __launch_bounds__(512)
void read_fused(const float* __restrict__ q, const float* __restrict__ v,
                const float* __restrict__ wrt, const float* __restrict__ entry,
                float* __restrict__ ctx) {
    extern __shared__ float smem[];
    float* st   = smem;                 // 32768 floats at LDS offset 0 (TDM target)
    float* qrow = st + Sn * Dm;         // 1024
    float* vrow = qrow + Dm;            // 1024
    float* wr   = vrow + Dm;            // 32
    float* sc   = wr + Sn;              // 32
    float* attn = sc + Sn;              // 32

    int b = blockIdx.x >> 4, c = blockIdx.x & 15;
    int tid = threadIdx.x, wave = tid >> 5, lane = tid & 31;

    // load chunk entry state (S*D fp32 = 128 KB)
#if HAVE_TDM
    const bool waveLeader = (__builtin_amdgcn_readfirstlane(wave) == 0);
    if (waveLeader) {
        // two 64 KB 1-D TDM copies (tile_dim0 is 16-bit)
        tdm_issue_f32_1d(0u, entry + (size_t)(b * NC + c) * Sn * Dm, 16384u);
        tdm_issue_f32_1d(65536u, entry + (size_t)(b * NC + c) * Sn * Dm + 16384, 16384u);
        __builtin_amdgcn_s_wait_tensorcnt(0);
    }
#else
    const float4* ge = (const float4*)(entry + (size_t)(b * NC + c) * Sn * Dm);
    float4* se = (float4*)st;
    #pragma unroll
    for (int j = 0; j < 16; ++j) se[j * 512 + tid] = ge[j * 512 + tid];
#endif
    __syncthreads();

    for (int i = 0; i < Cn; ++i) {
        int g = b * Tn + c * Cn + i;
        qrow[tid]       = q[(size_t)g * Dm + tid];
        qrow[512 + tid] = q[(size_t)g * Dm + 512 + tid];
        vrow[tid]       = v[(size_t)g * Dm + tid];
        vrow[512 + tid] = v[(size_t)g * Dm + 512 + tid];
        if (tid < Sn) wr[tid] = wrt[(size_t)g * Sn + tid];
        __syncthreads();

        // EMA state update: st = a*st + alpha*w[s]*v[d]
        #pragma unroll
        for (int j = 0; j < 64; ++j) {
            int idx = j * 512 + tid;
            int s = idx >> 10, d = idx & (Dm - 1);
            st[idx] = Af * st[idx] + ALPHA * wr[s] * vrow[d];
        }
        __syncthreads();

        // scores: 16 waves x 2 slots
        #pragma unroll
        for (int si = 0; si < 2; ++si) {
            int sl = wave * 2 + si;
            float p = 0.f;
            for (int d = lane; d < Dm; d += 32) p += qrow[d] * st[sl * Dm + d];
            p = wave_sum(p);
            if (lane == 0) sc[sl] = p * SCALE;
        }
        __syncthreads();

        if (tid < 32) {
            float x = sc[tid];
            float m = wave_max(x);
            float e = __expf(x - m);
            float sum = wave_sum(e);
            attn[tid] = e / sum;
        }
        __syncthreads();

        // ctx[d] = sum_s attn[s] * st[s,d]
        #pragma unroll
        for (int r = 0; r < 2; ++r) {
            int d = r * 512 + tid;
            float o = 0.f;
            #pragma unroll
            for (int s = 0; s < Sn; ++s) o += attn[s] * st[s * Dm + d];
            ctx[(size_t)g * Dm + d] = o;
        }
        __syncthreads();
    }
}

// ---------------- launcher ----------------
extern "C" void kernel_launch(void* const* d_in, const int* in_sizes, int n_in,
                              void* d_out, int out_size, void* d_ws, size_t ws_size,
                              hipStream_t stream) {
    const float* h      = (const float*)d_in[0];   // (B,T,D)
    const float* state0 = (const float*)d_in[1];   // (B,S,D)
    const float* gamma  = (const float*)d_in[2];
    const float* beta   = (const float*)d_in[3];
    const float* Wq     = (const float*)d_in[4];
    const float* bq     = (const float*)d_in[5];
    const float* Wv     = (const float*)d_in[6];
    const float* bv     = (const float*)d_in[7];

    float* ctx        = (float*)d_out;                          // (B,T,D)
    float* finalState = (float*)d_out + (size_t)Bn * Tn * Dm;   // (B,S,D)

    char* ws = (char*)d_ws;
    size_t off = 0;
    unsigned short* hn   = (unsigned short*)(ws + off); off += (size_t)Bn*Tn*Dm*2;   // bf16 h_norm
    unsigned short* wq16 = (unsigned short*)(ws + off); off += (size_t)Dm*Dm*2;
    unsigned short* wv16 = (unsigned short*)(ws + off); off += (size_t)Dm*Dm*2;
    float* qb    = (float*)(ws + off); off += (size_t)Bn*Tn*Dm*4;
    float* vb    = (float*)(ws + off); off += (size_t)Bn*Tn*Dm*4;
    float* wrt   = (float*)(ws + off); off += (size_t)Bn*Tn*Sn*4;
    float* local = (float*)(ws + off); off += (size_t)Bn*NC*Sn*Dm*4;
    float* entry = (float*)(ws + off); off += (size_t)Bn*NC*Sn*Dm*4;

    // 1. weight cast
    cast_weights<<<dim3(Dm*Dm/1024, 2), 256, 0, stream>>>(Wq, Wv, wq16, wv16);
    // 2. layernorm + bf16 cast
    ln_cast<<<Bn*Tn, 256, 0, stream>>>(h, gamma, beta, hn);
    // 3. WMMA projections (q and v), TDM double-buffered LDS staging
    gemm_qv<<<dim3(Bn*Tn/128, Dm/128, 2), 256, 2*BUFBYTES, stream>>>(
        hn, wq16, wv16, bq, bv, qb, vb);
    // 4. routing softmax
    routing<<<Bn*Tn, 128, 0, stream>>>(qb, state0, wrt);
    // 5. chunk-local decayed sums + carry scan (also writes final_state)
    chunk_local<<<Bn*NC, 1024, 0, stream>>>(vb, wrt, local);
    chunk_scan<<<Bn*128, 256, 0, stream>>>(state0, local, entry, finalState);
    // 6. fused replay + read (139.6 KB dynamic LDS per workgroup; 320 KB/WGP on CDNA5)
    size_t smem = (size_t)(Sn*Dm + 2*Dm + 3*Sn) * sizeof(float);
    read_fused<<<Bn*NC, 512, smem, stream>>>(qb, vb, wrt, entry, ctx);
}
